// NeighborhoodAttentionGeneric_46334107189545
// MI455X (gfx1250) — matrix operand order repaired and checked by reading, not compiled, not run on hardware
//
#include <hip/hip_runtime.h>

// ---------------------------------------------------------------------------
// CDNA5 (gfx1250) neighborhood attention, all matmuls via v_wmma_f32_16x16x32_bf16
// GEMM B-tiles staged via global_load_async_to_lds_b128 (ASYNCcnt, double buffer)
// ---------------------------------------------------------------------------

typedef __attribute__((ext_vector_type(16))) __bf16 v16bf;
typedef __attribute__((ext_vector_type(8)))  float  v8f;

union Frag { uint4 q[2]; v16bf v; };
union U8   { uint4 q; unsigned short e[8]; };

__device__ __forceinline__ unsigned short f2bf(float f) {
    unsigned u = __builtin_bit_cast(unsigned, f);
    return (unsigned short)((u + 0x7FFFu + ((u >> 16) & 1u)) >> 16);   // RNE
}

__device__ __forceinline__ v8f wmma_bf16(const Frag& a, const Frag& b, v8f c) {
    return __builtin_amdgcn_wmma_f32_16x16x32_bf16(false, a.v, false, b.v,
                                                   (short)0, c, false, false);
}

// ---- CDNA5 async global->LDS copy (tracked by ASYNCcnt) -------------------
__device__ __forceinline__ void async_ldsb128(unsigned lds_off,
                                              const unsigned short* g) {
    asm volatile("global_load_async_to_lds_b128 %0, %1, off"
                 :: "v"(lds_off), "v"(g) : "memory");
}
__device__ __forceinline__ void wait_async0() {
    asm volatile("s_wait_asynccnt 0x0" ::: "memory");
}

// --------------------------- fp32 -> bf16 (row major) ----------------------
__global__ void k_f32_to_bf16(const float* __restrict__ in,
                              unsigned short* __restrict__ out, int n4) {
    int i = blockIdx.x * blockDim.x + threadIdx.x;
    if (i >= n4) return;
    float4 f = ((const float4*)in)[i];
    ushort4 o;
    o.x = f2bf(f.x); o.y = f2bf(f.y); o.z = f2bf(f.z); o.w = f2bf(f.w);
    ((ushort4*)out)[i] = o;
}

// ------------------ pack W[K,N] fp32 into WMMA B-fragment order ------------
// fragment flat layout: (((nt*KT + kt)*32 + lane)*16 + i)
// element (lane,i): K = kt*32 + i + (lane&16), N = nt*16 + (lane&15)
__global__ void k_pack_b(const float* __restrict__ w,
                         unsigned short* __restrict__ pk, int K, int N) {
    int lane = threadIdx.x & 31;
    int cell = blockIdx.x * (blockDim.x >> 5) + (threadIdx.x >> 5);
    int KT = K >> 5, NT = N >> 4;
    if (cell >= NT * KT) return;
    int nt = cell / KT, kt = cell % KT;
    unsigned short* dst = pk + (size_t)cell * 512 + lane * 16;
    int n = nt * 16 + (lane & 15);
    int kbase = kt * 32 + (lane & 16);
#pragma unroll
    for (int i = 0; i < 16; ++i)
        dst[i] = f2bf(w[(size_t)(kbase + i) * N + n]);
}

// --------------------------- GEMM: C = A * Bpack + bias --------------------
// Block = 4 waves = 4 consecutive M-tiles sharing one 64-col group.
// B tile (4 frags = 4KB) is async-staged to LDS, double buffered.
template <bool OUT_F32>
__global__ __launch_bounds__(128)
void k_gemm(const unsigned short* __restrict__ A,
            const unsigned short* __restrict__ Bp,
            const float* __restrict__ bias, void* __restrict__ Out,
            int Ngrp, int KT, int N) {
    __shared__ unsigned short ldsB[2][2048];            // 2 x 4KB
    int tid  = threadIdx.x;
    int lane = tid & 31;
    int wave = tid >> 5;                                // mt within group
    int mg = blockIdx.x / Ngrp, ng = blockIdx.x % Ngrp;
    int mt = mg * 4 + wave;
    int K = KT * 32;
    int half8 = (lane & 16) >> 1;                       // 0 or 8
    const unsigned short* a0 = A + (size_t)(mt * 16 + (lane & 15)) * K + (lane & 16);
    unsigned ldsbase = (unsigned)(unsigned long long)(void*)&ldsB[0][0];

    auto issue = [&](int kt, int buf) {
#pragma unroll
        for (int it = 0; it < 2; ++it) {
            int flat = tid + it * 128;                  // 0..255 b128 chunks
            int j = flat >> 6, off = flat & 63;
            const unsigned short* src =
                Bp + (size_t)((ng * 4 + j) * KT + kt) * 512 + off * 8;
            async_ldsb128(ldsbase + buf * 4096 + flat * 16, src);
        }
    };

    issue(0, 0);
    v8f acc[4] = {};
    for (int kt = 0; kt < KT; ++kt) {
        wait_async0();                                  // my async writes done
        __syncthreads();                                // everyone's done
        if (kt + 1 < KT) issue(kt + 1, (kt + 1) & 1);   // prefetch next tile
        Frag af;
        af.q[0] = *(const uint4*)(a0 + kt * 32);        // K: (lane&16) + 0..7
        af.q[1] = *(const uint4*)(a0 + kt * 32 + 8);    // K: (lane&16) + 8..15
        const unsigned short* bb = &ldsB[kt & 1][0] + lane * 16;
#pragma unroll
        for (int j = 0; j < 4; ++j) {
            Frag bf;
            bf.q[0] = *(const uint4*)(bb + j * 512);
            bf.q[1] = *(const uint4*)(bb + j * 512 + 8);
            acc[j] = wmma_bf16(af, bf, acc[j]);
        }
    }
#pragma unroll
    for (int j = 0; j < 4; ++j) {
        int col = (ng * 4 + j) * 16 + (lane & 15);
        float bv = bias[col];
#pragma unroll
        for (int r = 0; r < 8; ++r) {
            int orow = mt * 16 + r + half8;
            float v = acc[j][r] + bv;
            if (OUT_F32) ((float*)Out)[(size_t)orow * N + col] = v;
            else ((unsigned short*)Out)[(size_t)orow * N + col] = f2bf(v);
        }
    }
}

// --------------------------- neighborhood attention ------------------------
// One block per 4x4 spatial tile (14x14 tiles), wave w = head w.
// Union of all 16 windows is a fixed 10x10 region (100 keys, pad to 112/128).
#define WW 56
#define CC 512
#define HD 64
#define VSTR 136   // v_t row stride (bf16): 272B -> 16B aligned, bank-spread
#define PSTR 136   // p row stride

__global__ __launch_bounds__(256)
void k_natten(const unsigned short* __restrict__ qkv,
              unsigned short* __restrict__ o) {
    __shared__ unsigned short v_t[512 * VSTR];          // V transposed [d][key]
    __shared__ unsigned short p_lds[8 * 16 * PSTR];     // per-wave P tiles

    int tid = threadIdx.x, lane = tid & 31, head = tid >> 5;
    int t = blockIdx.x;
    int h0 = (t / 14) * 4, w0 = (t % 14) * 4;
    int ush = h0 - 3; ush = ush < 0 ? 0 : (ush > 46 ? 46 : ush);
    int usw = w0 - 3; usw = usw < 0 ? 0 : (usw > 46 ? 46 : usw);

    // ---- stage V transposed into LDS: 100 keys x 512 d, 8 elems/thread/iter
    for (int it = 0; it < 25; ++it) {
        int idx = tid + it * 256;                       // 0..6399
        int kr  = idx >> 6;                             // key 0..99
        int d0  = (idx & 63) << 3;
        int kh  = ush + kr / 10, kw = usw + kr % 10;
        U8 u;
        u.q = *(const uint4*)(qkv + (size_t)(kh * WW + kw) * 3 * CC + 2 * CC + d0);
#pragma unroll
        for (int j = 0; j < 8; ++j) v_t[(d0 + j) * VSTR + kr] = u.e[j];
    }

    int colq  = lane & 15;
    int half8 = (lane & 16) >> 1;

    // ---- Q A-fragments (16 queries x 64d -> 2 frags of K=32)
    int qh = h0 + (colq >> 2), qw = w0 + (colq & 3);
    const unsigned short* qptr =
        qkv + (size_t)(qh * WW + qw) * 3 * CC + head * HD + (lane & 16);
    Frag qa[2];
#pragma unroll
    for (int kf = 0; kf < 2; ++kf) {
        qa[kf].q[0] = *(const uint4*)(qptr + kf * 32);
        qa[kf].q[1] = *(const uint4*)(qptr + kf * 32 + 8);
    }

    // ---- logits: 16q x 112key, 7 N-tiles x 2 K-frags
    v8f accl[7] = {};
#pragma unroll
    for (int nt = 0; nt < 7; ++nt) {
        int ki = nt * 16 + colq;
        int kd = ki / 10, km = ki - kd * 10;
        int krow = (ki < 100) ? ((ush + kd) * WW + (usw + km)) : 0;
        const unsigned short* kp =
            qkv + (size_t)krow * 3 * CC + CC + head * HD + (lane & 16);
#pragma unroll
        for (int kf = 0; kf < 2; ++kf) {
            Frag kb;
            kb.q[0] = *(const uint4*)(kp + kf * 32);
            kb.q[1] = *(const uint4*)(kp + kf * 32 + 8);
            accl[nt] = wmma_bf16(qa[kf], kb, accl[nt]);
        }
    }

    // ---- mask + softmax (row-wise over 112, cross-lane via shfl_xor)
    const float SCALE = 0.125f;                         // 64^-0.5
    int sh[8], sw[8];
#pragma unroll
    for (int r = 0; r < 8; ++r) {
        int row = r + half8;
        int h = h0 + (row >> 2), w = w0 + (row & 3);
        int s = h - 3; sh[r] = s < 0 ? 0 : (s > 49 ? 49 : s);
        s = w - 3;     sw[r] = s < 0 ? 0 : (s > 49 ? 49 : s);
    }
    float l[7][8];
#pragma unroll
    for (int nt = 0; nt < 7; ++nt) {
        int ki = nt * 16 + colq;
        int kd = ki / 10, km = ki - kd * 10;
        int kh = ush + kd, kw = usw + km;
        bool kvld = ki < 100;
#pragma unroll
        for (int r = 0; r < 8; ++r) {
            bool in = kvld && kh >= sh[r] && kh < sh[r] + 7 &&
                              kw >= sw[r] && kw < sw[r] + 7;
            l[nt][r] = in ? accl[nt][r] * SCALE : -3.0e38f;
        }
    }
#pragma unroll
    for (int r = 0; r < 8; ++r) {
        float m = l[0][r];
#pragma unroll
        for (int nt = 1; nt < 7; ++nt) m = fmaxf(m, l[nt][r]);
#pragma unroll
        for (int off = 1; off < 16; off <<= 1)
            m = fmaxf(m, __shfl_xor(m, off, 32));
        float s = 0.f;
#pragma unroll
        for (int nt = 0; nt < 7; ++nt) {
            float e = (l[nt][r] > -1.0e38f) ? __expf(l[nt][r] - m) : 0.f;
            l[nt][r] = e; s += e;
        }
#pragma unroll
        for (int off = 1; off < 16; off <<= 1) s += __shfl_xor(s, off, 32);
        float inv = 1.0f / s;
#pragma unroll
        for (int nt = 0; nt < 7; ++nt) l[nt][r] *= inv;
    }

    // ---- P: C-layout regs -> A-layout via LDS (bf16), zero-pad keys 112..127
    unsigned short* pw = p_lds + head * 16 * PSTR;
#pragma unroll
    for (int nt = 0; nt < 7; ++nt)
#pragma unroll
        for (int r = 0; r < 8; ++r)
            pw[(r + half8) * PSTR + nt * 16 + colq] = f2bf(l[nt][r]);
#pragma unroll
    for (int r = 0; r < 8; ++r)
        pw[(r + half8) * PSTR + 112 + colq] = 0;

    __syncthreads();                                    // v_t ready block-wide

    // ---- O = P @ V : 4 N-tiles (d) x 4 K-frags (keys 0..127, 112+ are zero)
    const unsigned short* pa0 = pw + colq * PSTR + (lane & 16);
#pragma unroll
    for (int nt2 = 0; nt2 < 4; ++nt2) {
        v8f acco = {};
        int col = head * HD + nt2 * 16 + colq;
        const unsigned short* vb0 = v_t + (size_t)col * VSTR + (lane & 16);
#pragma unroll
        for (int kf = 0; kf < 4; ++kf) {
            Frag pa, vb;
            pa.q[0] = *(const uint4*)(pa0 + kf * 32);
            pa.q[1] = *(const uint4*)(pa0 + kf * 32 + 8);
            vb.q[0] = *(const uint4*)(vb0 + kf * 32);
            vb.q[1] = *(const uint4*)(vb0 + kf * 32 + 8);
            acco = wmma_bf16(pa, vb, acco);
        }
#pragma unroll
        for (int r = 0; r < 8; ++r) {
            int row = r + half8;
            int pix = (h0 + (row >> 2)) * WW + w0 + (row & 3);
            o[(size_t)pix * CC + col] = f2bf(acco[r]);
        }
    }
}

// ---------------------------------------------------------------------------
extern "C" void kernel_launch(void* const* d_in, const int* in_sizes, int n_in,
                              void* d_out, int out_size, void* d_ws, size_t ws_size,
                              hipStream_t stream) {
    (void)in_sizes; (void)n_in; (void)out_size; (void)ws_size;
    const float* x      = (const float*)d_in[0];
    const float* w_qkv  = (const float*)d_in[1];
    const float* b_qkv  = (const float*)d_in[2];
    const float* w_proj = (const float*)d_in[3];
    const float* b_proj = (const float*)d_in[4];
    float* out = (float*)d_out;

    char* ws = (char*)d_ws;
    const size_t A256 = 255;
    size_t o0 = 0;
    unsigned short* x_bf   = (unsigned short*)(ws + o0);
    size_t o1 = o0 + (((size_t)3136 * 512  * 2 + A256) & ~A256);
    unsigned short* wqkv_p = (unsigned short*)(ws + o1);
    size_t o2 = o1 + (((size_t)512  * 1536 * 2 + A256) & ~A256);
    unsigned short* wprj_p = (unsigned short*)(ws + o2);
    size_t o3 = o2 + (((size_t)512  * 512  * 2 + A256) & ~A256);
    unsigned short* qkv_bf = (unsigned short*)(ws + o3);
    size_t o4 = o3 + (((size_t)3136 * 1536 * 2 + A256) & ~A256);
    unsigned short* o_bf   = (unsigned short*)(ws + o4);

    // 1) x -> bf16
    k_f32_to_bf16<<<(3136 * 512 / 4 + 255) / 256, 256, 0, stream>>>(
        x, x_bf, 3136 * 512 / 4);
    // 2) pack weights into WMMA B-fragment order
    k_pack_b<<<(96 * 16) / 8, 256, 0, stream>>>(w_qkv, wqkv_p, 512, 1536);
    k_pack_b<<<(32 * 16) / 8, 256, 0, stream>>>(w_proj, wprj_p, 512, 512);
    // 3) qkv = x @ w_qkv + b_qkv  (49 M-groups x 24 N-groups, 4 waves/block)
    k_gemm<false><<<49 * 24, 128, 0, stream>>>(x_bf, wqkv_p, b_qkv, qkv_bf,
                                               24, 16, 1536);
    // 4) neighborhood attention (196 tiles x 8 heads)
    k_natten<<<196, 256, 0, stream>>>(qkv_bf, o_bf);
    // 5) out = o @ w_proj + b_proj (49 x 8 blocks)
    k_gemm<true><<<49 * 8, 128, 0, stream>>>(o_bf, wprj_p, b_proj, out,
                                             8, 16, 512);
}
